// SymmetricContractionWrapper_6957847019586
// MI455X (gfx1250) — matrix operand order repaired and checked by reading, not compiled, not run on hardware
//
#include <hip/hip_runtime.h>
#include <hip/hip_bf16.h>

typedef __attribute__((ext_vector_type(16))) __bf16 v16bf;
typedef __attribute__((ext_vector_type(8)))  float  v8f;

#define NODES 4000
#define NCHAN 128
#define NELEM 10

// ---------------------------------------------------------------------------
// Kernel 1: one-hot attrs -> element index per node
// ---------------------------------------------------------------------------
__global__ void eidx_kernel(const float* __restrict__ attrs, int* __restrict__ eidx) {
    int b = blockIdx.x * 256 + threadIdx.x;
    if (b >= NODES) return;
    int e = 0; float best = -1.f;
    #pragma unroll
    for (int t = 0; t < NELEM; ++t) {
        float a = attrs[b * NELEM + t];
        if (a > best) { best = a; e = t; }
    }
    eidx[b] = e;
}

// ---------------------------------------------------------------------------
// K-ordering shared by pack kernel (B side) and A builder:
//   physical k -> h = (k>>3)&1 (lane-half bit), slot s = (k>>4)*8 + (k&7)
//   s <  9*HALF3      : p = s/HALF3, q = s%HALF3 + HALF3*h   (U3 / x_p*w3_q)
//   s <  9*HALF3 + 2  : q2 = (s - 9*HALF3) + 2*h             (U2 / w2_q2)
//   else              : zero padding
// For A-fragment elements, s = ks*16 + jj (lane-half independent).
// ---------------------------------------------------------------------------
__global__ void pack_u_kernel(const float* __restrict__ U3, const float* __restrict__ U2,
                              __bf16* __restrict__ Upk,
                              int K3, int K3P, int K2, int KS, int NCOLS, int total) {
    int idx = blockIdx.x * 256 + threadIdx.x;
    if (idx >= total) return;
    int j    = idx & 15;
    int lane = (idx >> 4) & 31;
    int frag = idx >> 9;
    int ks = frag % KS;
    int nt = frag / KS;
    int col = nt * 16 + (lane & 15);
    // WMMA bf16 B-operand: lane L -> col = nt*16 + L%16; k = ks*32 + (j<8?0:16) + (L<16?0:8) + (j&7)
    int k = ks * 32 + ((j & 8) ? 16 : 0) + ((lane & 16) ? 8 : 0) + (j & 7);
    int h = (k >> 3) & 1;
    int s = (k >> 4) * 8 + (k & 7);
    int HALF3 = K3P >> 1;
    float v = 0.f;
    if (col < NCOLS) {
        int d = col / 81, rem = col % 81, ii = rem / 9, jj = rem % 9;
        if (s < 9 * HALF3) {
            int p = s / HALF3;
            int q = s % HALF3 + HALF3 * h;
            if (q < K3) v = U3[(((d * 9 + ii) * 9 + jj) * 9 + p) * K3 + q];
        } else if (s < 9 * HALF3 + 2) {
            int q = (s - 9 * HALF3) + 2 * h;
            if (q < K2) v = U2[((d * 9 + ii) * 9 + jj) * K2 + q];
        }
    }
    Upk[idx] = (__bf16)v;
}

// ---------------------------------------------------------------------------
// Branch-free register-direct A-fragment builder (s = ks*16 + jj).
// ---------------------------------------------------------------------------
template<int HALF3, int KS>
__device__ __forceinline__ void build_afrag(v16bf* frag, const float* xv,
                                            const float* w3s, const float* w2s) {
    #pragma unroll
    for (int ks = 0; ks < KS; ++ks) {
        v16bf a;
        #pragma unroll
        for (int jj = 0; jj < 16; ++jj) {
            const int s = ks * 16 + jj;
            float v = 0.f;
            if (s < 9 * HALF3)          v = xv[s / HALF3] * w3s[s % HALF3];
            else if (s < 9 * HALF3 + 2) v = w2s[s - 9 * HALF3];
            a[jj] = (__bf16)v;
        }
        frag[ks] = a;
    }
}

// ---------------------------------------------------------------------------
// Kernel 3: main contraction. One block = one node = 128 channels = 4 waves.
// Each wave owns 32 channels (two 16-row A tiles -> 2 WMMAs per B load).
//   GEMM: c2[row, dij] = F[row, :] @ Uext[:, dij]   (bf16 WMMA, f32 acc)
//   Epilogue: out2[row,d,i] += c2[row,dij] * x_j    (ds_add_f32, imm offsets)
//             out[row,d]     = sum_i (out2 + U1*w1) * x_i
// ---------------------------------------------------------------------------
template<int D, int K3, int K2, int NT, int KS>
__global__ __launch_bounds__(128)
void contract_kernel(const float* __restrict__ x,
                     const int*   __restrict__ eidx,
                     const __bf16* __restrict__ Upk,
                     const float* __restrict__ U1,
                     const float* __restrict__ W3,
                     const float* __restrict__ W2,
                     const float* __restrict__ W1,
                     float* __restrict__ out, int out_off) {
    constexpr int NCOLS = D * 81;
    constexpr int C     = NCHAN;
    constexpr int UELE  = NT * KS * 512;
    constexpr int K3P   = (K3 + 1) & ~1;
    constexpr int HALF3 = K3P / 2;
    static_assert(9 * HALF3 + 2 <= 16 * KS, "K window too small");

    extern __shared__ char smem[];
    float*  x_t  = (float*)smem;                    // [9][128] transposed x
    float*  out2 = x_t + 9 * C;                     // 4 waves * 32 rows * 32
    __bf16* Ul   = (__bf16*)(out2 + 4 * 32 * 32);   // UELE bf16

    const int b    = blockIdx.x;
    const int tid  = threadIdx.x;
    const int wave = tid >> 5;
    const int lane = tid & 31;
    const int e    = eidx[b];

    // Stage x[b] transposed (x_t[p*128+c]) and packed U into LDS.
    for (int t = tid; t < 9 * C; t += 128) {
        int p = t / C, c = t % C;
        x_t[t] = x[b * C * 9 + c * 9 + p];
    }
    for (int t = tid; t < UELE / 8; t += 128)
        ((float4*)Ul)[t] = ((const float4*)Upk)[t];
    __syncthreads();

    // Per-wave region: channels c0..c0+31 (tile0: +0..15, tile1: +16..31).
    const int c0 = wave * 32;
    float* o2 = out2 + wave * 32 * 32;
    for (int t = lane; t < 32 * 32; t += 32) o2[t] = 0.f;

    // Per-lane row data: tile0 row r0 = lane%16 (channel cA), tile1 row r0+16.
    const int cA = c0 + (lane & 15);
    const int cB = cA + 16;
    const int h  = (lane >> 4) & 1;
    float xvA[9], xvB[9], w3sA[HALF3], w3sB[HALF3], w2sA[2], w2sB[2];
    #pragma unroll
    for (int p = 0; p < 9; ++p) { xvA[p] = x_t[p * C + cA]; xvB[p] = x_t[p * C + cB]; }
    #pragma unroll
    for (int t = 0; t < HALF3; ++t) {           // load only this half's weights
        int q = t + HALF3 * h;
        bool ok = q < K3;
        w3sA[t] = ok ? W3[(e * K3 + q) * C + cA] : 0.f;
        w3sB[t] = ok ? W3[(e * K3 + q) * C + cB] : 0.f;
    }
    #pragma unroll
    for (int t = 0; t < 2; ++t) {
        int q = t + 2 * h;
        bool ok = q < K2;
        w2sA[t] = ok ? W2[(e * K2 + q) * C + cA] : 0.f;
        w2sB[t] = ok ? W2[(e * K2 + q) * C + cB] : 0.f;
    }

    // Build A fragments entirely in registers, single path for both halves.
    v16bf afrag0[KS], afrag1[KS];
    build_afrag<HALF3, KS>(afrag0, xvA, w3sA, w2sA);
    build_afrag<HALF3, KS>(afrag1, xvB, w3sB, w2sB);

    // GEMM over column tiles + incremental j-contraction scatter.
    const int rbase = (lane & 16) ? 8 : 0;
    for (int nt = 0; nt < NT; ++nt) {
        v8f acc0 = {}, acc1 = {};
        #pragma unroll
        for (int ks = 0; ks < KS; ++ks) {
            v16bf bfrag = *(const v16bf*)(Ul + ((nt * KS + ks) * 32 + lane) * 16);
            acc0 = __builtin_amdgcn_wmma_f32_16x16x32_bf16(
                false, afrag0[ks], false, bfrag, (short)0, acc0, false, false);
            acc1 = __builtin_amdgcn_wmma_f32_16x16x32_bf16(
                false, afrag1[ks], false, bfrag, (short)0, acc1, false, false);
        }
        // D layout: VGPR v, lanes0-15 -> row v, lanes16-31 -> row v+8, col = lane%16.
        int col = nt * 16 + (lane & 15);
        if (col < NCOLS) {
            int d = col / 81, rem = col % 81, i = rem / 9, j = rem % 9;
            int di = d * 9 + i;
            // 8 contiguous x values per tile (transposed layout) -> b128 loads.
            const float* xb = x_t + j * C + c0 + rbase;
            float4 xl0 = *(const float4*)(xb);
            float4 xh0 = *(const float4*)(xb + 4);
            float4 xl1 = *(const float4*)(xb + 16);
            float4 xh1 = *(const float4*)(xb + 20);
            float xr0[8] = {xl0.x, xl0.y, xl0.z, xl0.w, xh0.x, xh0.y, xh0.z, xh0.w};
            float xr1[8] = {xl1.x, xl1.y, xl1.z, xl1.w, xh1.x, xh1.y, xh1.z, xh1.w};
            float* o2b0 = o2 + rbase * 32 + di;           // + v*32 (imm offset)
            float* o2b1 = o2b0 + 16 * 32;
            #pragma unroll
            for (int v = 0; v < 8; ++v)
                atomicAdd(o2b0 + v * 32, acc0[v] * xr0[v]);
            #pragma unroll
            for (int v = 0; v < 8; ++v)
                atomicAdd(o2b1 + v * 32, acc1[v] * xr1[v]);
        }
    }
    __syncthreads();

    // Final per-channel epilogue: c1 = out2 + U1*w1 ; out = sum_i c1 * x_i.
    {
        int c = c0 + lane;
        float w1c = W1[e * C + c];
        const float* o2r = o2 + lane * 32;
        #pragma unroll
        for (int d = 0; d < D; ++d) {
            float s = 0.f;
            #pragma unroll
            for (int i = 0; i < 9; ++i) {
                float c1 = o2r[d * 9 + i] + U1[d * 9 + i] * w1c;
                s += c1 * x_t[i * C + c];
            }
            out[b * 512 + out_off + c * D + d] = s;
        }
    }
}

// ---------------------------------------------------------------------------
extern "C" void kernel_launch(void* const* d_in, const int* in_sizes, int n_in,
                              void* d_out, int out_size, void* d_ws, size_t ws_size,
                              hipStream_t stream) {
    const float* x     = (const float*)d_in[0];
    const float* attrs = (const float*)d_in[1];
    const float* U3_0  = (const float*)d_in[2];
    const float* U2_0  = (const float*)d_in[3];
    const float* U1_0  = (const float*)d_in[4];
    const float* W3_0  = (const float*)d_in[5];
    const float* W2_0  = (const float*)d_in[6];
    const float* W1_0  = (const float*)d_in[7];
    const float* U3_1  = (const float*)d_in[8];
    const float* U2_1  = (const float*)d_in[9];
    const float* U1_1  = (const float*)d_in[10];
    const float* W3_1  = (const float*)d_in[11];
    const float* W2_1  = (const float*)d_in[12];
    const float* W1_1  = (const float*)d_in[13];
    float* out = (float*)d_out;

    // Workspace: eidx (16 KB) | Upk0 (18.4 KB) | Upk1 (64 KB)
    int*    eidx = (int*)d_ws;
    __bf16* Upk0 = (__bf16*)((char*)d_ws + 16384);
    __bf16* Upk1 = (__bf16*)((char*)d_ws + 16384 + 20480);

    eidx_kernel<<<(NODES + 255) / 256, 256, 0, stream>>>(attrs, eidx);

    // irrep 0: D=1,K3=8 (K3P=8),K2=3 -> KS=3 (KPAD 96),  N=81  (NT=6)
    // irrep 1: D=3,K3=11(K3P=12),K2=4 -> KS=4 (KPAD 128), N=243 (NT=16)
    const int tot0 = 6 * 3 * 512, tot1 = 16 * 4 * 512;
    pack_u_kernel<<<(tot0 + 255) / 256, 256, 0, stream>>>(U3_0, U2_0, Upk0, 8, 8, 3, 3, 81, tot0);
    pack_u_kernel<<<(tot1 + 255) / 256, 256, 0, stream>>>(U3_1, U2_1, Upk1, 11, 12, 4, 4, 243, tot1);

    size_t smem0 = (size_t)(9 * NCHAN + 4 * 32 * 32) * 4 + (size_t)tot0 * 2;
    size_t smem1 = (size_t)(9 * NCHAN + 4 * 32 * 32) * 4 + (size_t)tot1 * 2;

    contract_kernel<1, 8, 3, 6, 3><<<NODES, 128, smem0, stream>>>(
        x, eidx, Upk0, U1_0, W3_0, W2_0, W1_0, out, 0);
    contract_kernel<3, 11, 4, 16, 4><<<NODES, 128, smem1, stream>>>(
        x, eidx, Upk1, U1_1, W3_1, W2_1, W1_1, out, 128);
}
